// Encoder_76630806495266
// MI455X (gfx1250) — matrix operand order repaired and checked by reading, not compiled, not run on hardware
//
#include <hip/hip_runtime.h>
#include <hip/hip_bf16.h>

// ---------------------------------------------------------------------------
// Problem constants
// ---------------------------------------------------------------------------
#define B_    128
#define T_    50
#define N_    24
#define D_    6
#define H_    256
#define G_    1024        // 4*H
#define O_    256
#define BM    16          // batch rows per block
#define NB    (B_ / BM)   // 8 batch chunks
#define KT_   9           // K-tiles for recurrent GEMM: 8 for h (256) + 1 for [x|0]
#define KE_   288         // extended K = 256 + 32
#define LDSTR 296         // padded bf16 row stride for [h | x | 0] in LDS (mult of 8)

typedef __bf16 v16bf __attribute__((ext_vector_type(16)));
typedef __bf16 v8bf  __attribute__((ext_vector_type(8)));
typedef float  v8f   __attribute__((ext_vector_type(8)));

static_assert(sizeof(v16bf) == 32, "v16bf must be 32 bytes");

// ---------------------------------------------------------------------------
// Workspace layout (bytes). Pre-swizzled bf16 WMMA B-fragments + gathered
// per-node fp32 bias tensors. Total ~17.4 MB (L2 resident on MI455X).
// ---------------------------------------------------------------------------
#define WS_WHHB  0ull                          // bf16 [N][9 kt][64 ntile][32 lane][16]
#define WS_WHHB_SZ (24ull * KT_ * 64 * 512 * 2)   // 14,155,776
#define WS_FCB   (WS_WHHB + WS_WHHB_SZ)        // bf16 [N][8 kt][16 ntile][32 lane][16]
#define WS_FCB_SZ  (24ull * 8 * 16 * 512 * 2)  // 3,145,728
#define WS_BIAS  (WS_FCB + WS_FCB_SZ)          // f32 [N][1024]  (b_ih+b_hh)
#define WS_BIAS_SZ (24ull * 1024 * 4)
#define WS_FCBI  (WS_BIAS + WS_BIAS_SZ)        // f32 [N][256]
#define WS_FCBI_SZ (24ull * 256 * 4)

__device__ __forceinline__ float sigmoidf_(float x) {
    return 1.0f / (1.0f + __expf(-x));
}
__device__ __forceinline__ float leakyf_(float x) {
    return x < 0.0f ? 0.01f * x : x;
}

// Load one 16x32 bf16 A-fragment from LDS [m][k] buffer.
// Contiguous k order used here (B prep matches): lanes 0-15: elems 0..15 -> K+0..15
//                          lanes16-31: elems 0..15 -> K+16..31
__device__ __forceinline__ v16bf load_a_frag(const __bf16* shh, int lane, int kt) {
    const __bf16* rp = shh + (lane & 15) * LDSTR + kt * 32 + ((lane >= 16) ? 16 : 0);
    v8bf lo = *(const v8bf*)rp;          // 16B aligned
    v8bf hi = *(const v8bf*)(rp + 8);    // 16B aligned
    return __builtin_shufflevector(lo, hi, 0, 1, 2, 3, 4, 5, 6, 7,
                                           8, 9, 10, 11, 12, 13, 14, 15);
}

// ---------------------------------------------------------------------------
// Pre-pass: gather per-node weights into pre-swizzled bf16 B-fragments
// (B 32x16, contiguous k order as in A: lanes 0-15 hold K=base+e, lanes 16-31 K=base+16+e,
// column = lane&15). K-tile 8 of the recurrent operand holds Wih rows
// (K=256..261 -> d=0..5) and zeros for K=262..287.
// ---------------------------------------------------------------------------
__global__ __launch_bounds__(256)
void prep_kernel(const int* __restrict__ node_types,
                 const float* __restrict__ w_ih, const float* __restrict__ w_hh,
                 const float* __restrict__ b_ih, const float* __restrict__ b_hh,
                 const float* __restrict__ fc_w, const float* __restrict__ fc_b,
                 char* __restrict__ ws)
{
    __bf16* whhb = (__bf16*)(ws + WS_WHHB);
    __bf16* fcbf = (__bf16*)(ws + WS_FCB);
    float*  biasg = (float*)(ws + WS_BIAS);
    float*  fcbi = (float*)(ws + WS_FCBI);

    const unsigned E0 = 24u * KT_ * 64 * 512; // Whh|Wih fragments (7,077,888)
    const unsigned E1 = 24u * 8 * 16 * 512;   // fc fragments      (1,572,864)
    const unsigned E2 = 24u * G_;             // bias
    const unsigned E3 = 24u * O_;             // fc bias
    const unsigned total = E0 + E1 + E2 + E3;

    for (unsigned idx = blockIdx.x * blockDim.x + threadIdx.x; idx < total;
         idx += gridDim.x * blockDim.x) {
        if (idx < E0) {
            unsigned f = idx >> 9, r = idx & 511;
            unsigned lane = r >> 4, e = r & 15;
            unsigned ntile = f & 63, tmp = f >> 6;
            unsigned kt = tmp % KT_, node = tmp / KT_;
            int nt = node_types[node];
            unsigned K = kt * 32 + ((lane >= 16) ? 16 : 0) + e;
            unsigned nout = ntile * 16 + (lane & 15);
            float v;
            if (kt < 8) {
                v = w_hh[((unsigned)nt * H_ + K) * G_ + nout];
            } else {
                unsigned d = K - 256;
                v = (d < D_) ? w_ih[((unsigned)nt * D_ + d) * G_ + nout] : 0.0f;
            }
            whhb[idx] = (__bf16)v;
        } else if (idx < E0 + E1) {
            unsigned j = idx - E0;
            unsigned f = j >> 9, r = j & 511;
            unsigned lane = r >> 4, e = r & 15;
            unsigned ntile = f & 15, kt = (f >> 4) & 7, node = f >> 7;
            int nt = node_types[node];
            unsigned K = kt * 32 + ((lane >= 16) ? 16 : 0) + e;
            unsigned nout = ntile * 16 + (lane & 15);
            fcbf[j] = (__bf16)fc_w[((unsigned)nt * H_ + K) * O_ + nout];
        } else if (idx < E0 + E1 + E2) {
            unsigned j = idx - E0 - E1;
            unsigned node = j >> 10, g = j & 1023;
            int nt = node_types[node];
            biasg[j] = b_ih[(unsigned)nt * G_ + g] + b_hh[(unsigned)nt * G_ + g];
        } else {
            unsigned j = idx - E0 - E1 - E2;
            unsigned node = j >> 8, g = j & 255;
            int nt = node_types[node];
            fcbi[j] = fc_b[(unsigned)nt * O_ + g];
        }
    }
}

// ---------------------------------------------------------------------------
// Main kernel: one block = 16 batch rows of one node, 8 waves (wave32).
// Wave w owns hidden columns [32w, 32w+32) of all 4 gates; c-state stays in
// its D-fragment registers. phase(k)=k/32+1 -> time mask uniform per wave.
// Gates = [h_t | x_t | 0] (16x288 bf16) @ [Whh ; Wih ; 0] via 72 WMMAs/wave.
// ---------------------------------------------------------------------------
__global__ __launch_bounds__(256)
void lstm_main(const float* __restrict__ x, float* __restrict__ out,
               const char* __restrict__ ws,
               const int* __restrict__ node_types,
               const float* __restrict__ h1_w, const float* __restrict__ h1_b,
               const float* __restrict__ h2_w, const float* __restrict__ h2_b)
{
    __shared__ __bf16 sh_h[BM * LDSTR];   // [m][0..255]=h_t, [256..261]=x_t, rest 0
    __shared__ float  sh_bias[G_];        // combined bias
    __shared__ float  sh_x[BM * 8];       // fp32 x0 tile (for exact h0/c0)

    const int tid  = threadIdx.x;
    const int lane = tid & 31;
    const int w    = tid >> 5;                 // wave id, 0..7
    const int node = blockIdx.x >> 3;
    const int b0   = (blockIdx.x & 7) * BM;
    const int nt   = node_types[node];
    const int ncol = lane & 15;
    const int mh   = (lane >= 16) ? 8 : 0;     // C/D row offset for high lanes

    const __bf16* whhb = (const __bf16*)(ws + WS_WHHB) + (size_t)node * KT_ * 64 * 512;
    const __bf16* fcbf = (const __bf16*)(ws + WS_FCB) + (size_t)node * 8 * 16 * 512;
    const float*  biasg = (const float*)(ws + WS_BIAS) + node * G_;
    const float*  fcbi = (const float*)(ws + WS_FCBI) + node * O_;

    // ---- one-time block setup ------------------------------------------------
    for (int i = tid; i < G_; i += 256) sh_bias[i] = biasg[i];
    for (int i = tid; i < BM * 32; i += 256) {       // zero the [x|pad] columns
        int m = i >> 5, c = i & 31;
        sh_h[m * LDSTR + 256 + c] = (__bf16)0.0f;
    }
    if (tid < BM * D_) {                             // stage fp32 x(t=0)
        int m = tid / D_, d = tid - m * D_;
        sh_x[m * 8 + d] = x[(((size_t)(b0 + m) * T_ + 0) * N_ + node) * D_ + d];
    }
    __syncthreads();

    // h0 = x0 @ h1_w + h1_b  (fp32, cooperative) -> sh_h cols 0..255 (bf16)
    for (int i = tid; i < BM * H_; i += 256) {
        int m = i >> 8, k = i & 255;
        float v = h1_b[nt * H_ + k];
#pragma unroll
        for (int d = 0; d < D_; d++) v += sh_x[m * 8 + d] * h1_w[(nt * D_ + d) * H_ + k];
        sh_h[m * LDSTR + k] = (__bf16)v;
    }
    if (tid < BM * D_) {                             // x0 -> extended K columns
        int m = tid / D_, d = tid - m * D_;
        sh_h[m * LDSTR + 256 + d] = (__bf16)sh_x[m * 8 + d];
    }
    // c0 = x0 @ h2_w + h2_b, directly in this wave's fragment layout (fp32 exact)
    float cst[2][8];
#pragma unroll
    for (int s = 0; s < 2; s++) {
        int hidx = (2 * w + s) * 16 + ncol;
        float bb = h2_b[nt * H_ + hidx];
        float wv[D_];
#pragma unroll
        for (int d = 0; d < D_; d++) wv[d] = h2_w[(nt * D_ + d) * H_ + hidx];
#pragma unroll
        for (int r = 0; r < 8; r++) {
            float v = bb;
#pragma unroll
            for (int d = 0; d < D_; d++) v += wv[d] * sh_x[(r + mh) * 8 + d];
            cst[s][r] = v;
        }
    }
    __syncthreads();

    // Hoist the 8 per-tile bias scalars into registers (loop-invariant).
    float bb8[8];
#pragma unroll
    for (int j = 0; j < 8; j++) {
        int q = j >> 1, s = j & 1;
        bb8[j] = sh_bias[q * 256 + (2 * w + s) * 16 + ncol];
    }

    float hlast[2][8] = {};

    // ---- time loop -----------------------------------------------------------
    for (int t = 0; t < T_; t++) {
        // Opaque zero offset: re-materialized every iteration so the compiler
        // cannot hoist the 72 B-fragment loads out of the loop (which would
        // spill 576 VGPRs to per-wave scratch and defeat L2 sharing of ws).
        int zoff = 0;
        asm volatile("" : "+v"(zoff));
        const __bf16* whhb_t = whhb + zoff;

        // A fragments: [h_t | x_t | 0], 9 K-tiles
        v16bf afrag[KT_];
#pragma unroll
        for (int kt = 0; kt < KT_; kt++) afrag[kt] = load_a_frag(sh_h, lane, kt);

        // 8 tiles: j = 2*q + s ; gate q, hidden tile ht = 2*w + s
        v8f acc[8];
#pragma unroll
        for (int j = 0; j < 8; j++) {
            float bb = bb8[j];
            acc[j] = (v8f){bb, bb, bb, bb, bb, bb, bb, bb};
        }
#pragma unroll
        for (int j = 0; j < 8; j++) {
            int q = j >> 1, s = j & 1;
            int ntile = q * 16 + 2 * w + s;                 // 0..63
            const __bf16* bbase = whhb_t + (size_t)ntile * 512 + lane * 16;
            if (j < 7)  // prefetch next tile's fragment stream
                __builtin_prefetch(whhb_t + (size_t)(((j + 1) >> 1) * 16 + 2 * w + ((j + 1) & 1)) * 512, 0, 3);
            v8f a = acc[j];
#pragma unroll
            for (int kt = 0; kt < KT_; kt++) {
                v16bf bf = *(const v16bf*)(bbase + (size_t)kt * 64 * 512);
                a = __builtin_amdgcn_wmma_f32_16x16x32_bf16(
                        false, afrag[kt], false, bf, (short)0, a, false, false);
            }
            acc[j] = a;
        }

        // cell update; phase = w+1 -> mask uniform per wave
        const bool cm = (t % (w + 1)) == 0;
#pragma unroll
        for (int s = 0; s < 2; s++) {
#pragma unroll
            for (int r = 0; r < 8; r++) {
                float iv = sigmoidf_(acc[0 + s][r]);
                float fv = sigmoidf_(acc[2 + s][r]);
                float gv = tanhf(acc[4 + s][r]);
                float ov = sigmoidf_(acc[6 + s][r]);
                float c  = cst[s][r];
                c = cm ? (fv * c + iv * gv) : c;
                cst[s][r] = c;
                hlast[s][r] = ov * tanhf(c);
            }
        }
        __syncthreads();   // all sh_h reads of this step done
        // publish h_{t+1} (bf16) and stage x_{t+1} into extended K columns
#pragma unroll
        for (int s = 0; s < 2; s++) {
            int hidx = (2 * w + s) * 16 + ncol;
#pragma unroll
            for (int r = 0; r < 8; r++)
                sh_h[(r + mh) * LDSTR + hidx] = (__bf16)hlast[s][r];
        }
        if (t + 1 < T_ && tid < BM * D_) {
            int m = tid / D_, d = tid - m * D_;
            sh_h[m * LDSTR + 256 + d] =
                (__bf16)x[(((size_t)(b0 + m) * T_ + (t + 1)) * N_ + node) * D_ + d];
        }
        __syncthreads();
    }

    // ---- store hT, cT, leaky(y_last) (y_last == hT) --------------------------
    const size_t off_hT = 786432, off_cT = 1572864, off_lk = 2359296;
#pragma unroll
    for (int s = 0; s < 2; s++) {
        int hidx = (2 * w + s) * 16 + ncol;
#pragma unroll
        for (int r = 0; r < 8; r++) {
            size_t base = ((size_t)(b0 + r + mh) * N_ + node) * H_ + hidx;
            float hy = hlast[s][r];
            out[off_hT + base] = hy;
            out[off_cT + base] = cst[s][r];
            out[off_lk + base] = leakyf_(hy);
        }
    }

    // ---- epilogue FC: h_out = leaky(hT @ fc_w + fc_b), K-tiles 0..7 only -----
    v16bf a2[8];
#pragma unroll
    for (int kt = 0; kt < 8; kt++) a2[kt] = load_a_frag(sh_h, lane, kt);
#pragma unroll
    for (int s = 0; s < 2; s++) {
        int tile = 2 * w + s;                  // O-tile 0..15
        int nout = tile * 16 + ncol;
        float bb = fcbi[nout];
        v8f a = {bb, bb, bb, bb, bb, bb, bb, bb};
#pragma unroll
        for (int kt = 0; kt < 8; kt++) {
            v16bf bf = *(const v16bf*)(fcbf + ((size_t)kt * 16 + tile) * 512 + lane * 16);
            a = __builtin_amdgcn_wmma_f32_16x16x32_bf16(
                    false, a2[kt], false, bf, (short)0, a, false, false);
        }
#pragma unroll
        for (int r = 0; r < 8; r++) {
            out[((size_t)(b0 + r + mh) * N_ + node) * O_ + nout] = leakyf_(a[r]);
        }
    }
}

// ---------------------------------------------------------------------------
extern "C" void kernel_launch(void* const* d_in, const int* in_sizes, int n_in,
                              void* d_out, int out_size, void* d_ws, size_t ws_size,
                              hipStream_t stream) {
    const float* x          = (const float*)d_in[0];
    const int*   node_types = (const int*)d_in[1];
    const float* w_ih       = (const float*)d_in[2];
    const float* w_hh       = (const float*)d_in[3];
    const float* b_ih       = (const float*)d_in[4];
    const float* b_hh       = (const float*)d_in[5];
    const float* fc_w       = (const float*)d_in[6];
    const float* fc_b       = (const float*)d_in[7];
    const float* h1_w       = (const float*)d_in[8];
    const float* h1_b       = (const float*)d_in[9];
    const float* h2_w       = (const float*)d_in[10];
    const float* h2_b       = (const float*)d_in[11];
    float* out = (float*)d_out;
    char*  ws  = (char*)d_ws;

    prep_kernel<<<4096, 256, 0, stream>>>(node_types, w_ih, w_hh, b_ih, b_hh,
                                          fc_w, fc_b, ws);
    lstm_main<<<N_ * NB, 256, 0, stream>>>(x, out, ws, node_types,
                                           h1_w, h1_b, h2_w, h2_b);
}
